// TemporalGATTransformer_33612414059030
// MI455X (gfx1250) — compile-verified, matrix operand order checked by reading, not run on hardware
//
#include <hip/hip_runtime.h>
#include <hip/hip_bf16.h>
#include <math.h>

#define NP 10000
#define NB 50000
#define NE 300000
#define NS 4096
#define GH 4      // GAT heads
#define HC 64     // head channels
#define GD 256    // GH*HC
#define MD 128
#define LBH 5
#define NCL 8

typedef __attribute__((ext_vector_type(16))) __bf16 v16bf;
typedef __attribute__((ext_vector_type(8)))  __bf16 v8bf;
typedef __attribute__((ext_vector_type(8)))  float  v8f;

__device__ __forceinline__ __bf16 f2bf(float f) {
  unsigned u = __builtin_bit_cast(unsigned, f);
  unsigned r = u + 0x7FFFu + ((u >> 16) & 1u);     // round-to-nearest-even
  unsigned short h = (unsigned short)(r >> 16);
  return __builtin_bit_cast(__bf16, h);
}

__device__ __forceinline__ __bf16 bf16_zero() {
  return __builtin_bit_cast(__bf16, (unsigned short)0);
}

__device__ __forceinline__ float gelu_exact(float x) {
  return 0.5f * x * (1.0f + erff(x * 0.70710678118654752440f));
}

__device__ __forceinline__ void atomicMaxF(float* addr, float v) {
  if (v >= 0.0f) atomicMax((int*)addr, __float_as_int(v));
  else           atomicMin((unsigned int*)addr, __float_as_uint(v));
}

// Async global->LDS copy of 16B per active lane (gfx1250 VGLOBAL op 98).
// ldsByteAddr: group-relative LDS byte offset (low 32 bits of flat LDS ptr).
__device__ __forceinline__ void async_copy_b128(unsigned ldsByteAddr, const void* gaddr) {
  asm volatile("global_load_async_to_lds_b128 %0, %1, off"
               :: "v"(ldsByteAddr), "v"(gaddr) : "memory");
}
__device__ __forceinline__ void async_wait0() {
  asm volatile("s_wait_asynccnt 0x0" ::: "memory");
}

// ---------------------------------------------------------------------------
// f32 -> bf16 conversion (one-time per launch for GEMM-feeding tensors)
// ---------------------------------------------------------------------------
__global__ void f32_to_bf16_kernel(const float* __restrict__ in,
                                   __bf16* __restrict__ out, long long n) {
  long long i = (long long)blockIdx.x * blockDim.x + threadIdx.x;
  if (i < n) out[i] = f2bf(in[i]);
}

// ---------------------------------------------------------------------------
// Generic GEMM: C = act(A[M,K] @ W[N,K]^T + bias); A,W bf16; f32 WMMA acc.
// One wave -> one 16x16 tile. wave32 fragment layout per CDNA5 ISA:
//   kbase = (lane<16 ? 0 : 8); element e -> k = kbase + (e<8 ? e : e+8)
// so each operand fragment is two contiguous 16B (8 x bf16) loads.
// FULLK=true: K % 32 == 0, branch-free inner loop.
// Output: f32 (Cf) or bf16 (Cb), exactly one non-null.
// ---------------------------------------------------------------------------
template <bool FULLK>
__global__ __launch_bounds__(256)
void gemm_wmma_bf16(const __bf16* __restrict__ A, const __bf16* __restrict__ W,
                    const float* __restrict__ bias,
                    float* __restrict__ Cf, __bf16* __restrict__ Cb,
                    int M, int N, int K, int ldc, int act)
{
  const int lane = threadIdx.x & 31;
  const int wv   = threadIdx.x >> 5;
  const int tilesN = N >> 4;
  const int tilesM = (M + 15) >> 4;
  const int tile = blockIdx.x * 8 + wv;
  if (tile >= tilesM * tilesN) return;
  const int tm = tile / tilesN;
  const int tn = tile - tm * tilesN;
  const int hsel = lane >> 4;          // 0: lanes 0-15, 1: lanes 16-31
  const int r    = lane & 15;
  const int koff = hsel * 8;
  const int arow = tm * 16 + r;
  const int brow = tn * 16 + r;
  const __bf16* Ar = A + (size_t)(arow < M ? arow : 0) * K;
  const __bf16* Wr = W + (size_t)brow * K;

  v8f acc = {0.f, 0.f, 0.f, 0.f, 0.f, 0.f, 0.f, 0.f};

  for (int k0 = 0; k0 < K; k0 += 32) {
    v16bf a, b;
    if (FULLK || k0 + 32 <= K) {
      const v8bf alo = *(const v8bf*)(Ar + k0 + koff);
      const v8bf ahi = *(const v8bf*)(Ar + k0 + koff + 16);
      const v8bf blo = *(const v8bf*)(Wr + k0 + koff);
      const v8bf bhi = *(const v8bf*)(Wr + k0 + koff + 16);
      a = __builtin_shufflevector(alo, ahi, 0,1,2,3,4,5,6,7,8,9,10,11,12,13,14,15);
      b = __builtin_shufflevector(blo, bhi, 0,1,2,3,4,5,6,7,8,9,10,11,12,13,14,15);
      if (k0 + 32 < K) {
        __builtin_prefetch(Ar + k0 + 32, 0, 1);
        __builtin_prefetch(Wr + k0 + 32, 0, 1);
      }
    } else {
#pragma unroll
      for (int e = 0; e < 16; ++e) {
        const int k = k0 + koff + (e < 8 ? e : e + 8);
        a[e] = (k < K) ? Ar[k] : bf16_zero();
        b[e] = (k < K) ? Wr[k] : bf16_zero();
      }
    }
    acc = __builtin_amdgcn_wmma_f32_16x16x32_bf16(
        false, a, false, b, (short)0, acc, false, false);
  }

  const int col = tn * 16 + r;
  const float bv = bias ? bias[col] : 0.0f;
#pragma unroll
  for (int i = 0; i < 8; ++i) {
    const int row = tm * 16 + hsel * 8 + i;   // C layout: VGPR i -> M = i (+8 hi half)
    if (row < M) {
      float v = acc[i] + bv;
      if (act) v = gelu_exact(v);
      if (Cb) Cb[(size_t)row * ldc + col] = f2bf(v);
      else    Cf[(size_t)row * ldc + col] = v;
    }
  }
}

// ---------------------------------------------------------------------------
__global__ void fill_kernel(float* __restrict__ p, float v, long long n) {
  long long i = (long long)blockIdx.x * blockDim.x + threadIdx.x;
  if (i < n) p[i] = v;
}

// ---------------------------------------------------------------------------
// GATv2 edge logits: one wave per edge. em = ea[e] @ We^T computed on the fly
// from LDS-resident We (16KB, staged with async global->LDS DMA).
// logit[e][h] = sum_c lrelu(m)*att[h][c]; 8-lane head-group reduction via
// shfl_xor; atomic segment max into mx.
// ---------------------------------------------------------------------------
__global__ __launch_bounds__(256)
void edge_logits_kernel(const float* __restrict__ xl, const float* __restrict__ xr,
                        const float* __restrict__ ea, const float* __restrict__ We,
                        const float* __restrict__ attw,
                        const int* __restrict__ sidx, const int* __restrict__ didx,
                        float* __restrict__ logits, float* __restrict__ mx, int E)
{
  __shared__ __align__(16) float sWe[GD * 16];
  __shared__ __align__(16) float sAtt[GD];
  {
    // async DMA: each lane moves 16B; 256 threads x 4 passes = 16KB (sWe)
    const int t = threadIdx.x;
#pragma unroll
    for (int p2 = 0; p2 < 4; ++p2) {
      const int fi = (p2 * 256 + t) * 4;
      async_copy_b128((unsigned)(size_t)&sWe[fi], We + fi);
    }
    if (t < 64) {
      async_copy_b128((unsigned)(size_t)&sAtt[t * 4], attw + t * 4);
    }
    async_wait0();
  }
  __syncthreads();

  const int lane = threadIdx.x & 31;
  const int e = blockIdx.x * 8 + (threadIdx.x >> 5);
  if (e >= E) return;
  const int s = sidx[e], d = didx[e];

  float eav[16];
  {
    const float4* ep = (const float4*)(ea + (size_t)e * 16);
    float4 t0 = ep[0], t1 = ep[1], t2 = ep[2], t3 = ep[3];
    eav[0]=t0.x; eav[1]=t0.y; eav[2]=t0.z; eav[3]=t0.w;
    eav[4]=t1.x; eav[5]=t1.y; eav[6]=t1.z; eav[7]=t1.w;
    eav[8]=t2.x; eav[9]=t2.y; eav[10]=t2.z; eav[11]=t2.w;
    eav[12]=t3.x; eav[13]=t3.y; eav[14]=t3.z; eav[15]=t3.w;
  }
  const int c0 = lane * 8;
  float xlv[8], xrv[8];
  {
    const float4* lp = (const float4*)(xl + (size_t)s * GD + c0);
    float4 l0 = lp[0], l1 = lp[1];
    xlv[0]=l0.x; xlv[1]=l0.y; xlv[2]=l0.z; xlv[3]=l0.w;
    xlv[4]=l1.x; xlv[5]=l1.y; xlv[6]=l1.z; xlv[7]=l1.w;
    const float4* rp = (const float4*)(xr + (size_t)d * GD + c0);
    float4 r0 = rp[0], r1 = rp[1];
    xrv[0]=r0.x; xrv[1]=r0.y; xrv[2]=r0.z; xrv[3]=r0.w;
    xrv[4]=r1.x; xrv[5]=r1.y; xrv[6]=r1.z; xrv[7]=r1.w;
  }
  float partial = 0.0f;
#pragma unroll
  for (int j = 0; j < 8; ++j) {
    const float* w = &sWe[(c0 + j) * 16];
    float m = xlv[j] + xrv[j];
#pragma unroll
    for (int k = 0; k < 16; ++k) m += eav[k] * w[k];
    const float lr = (m > 0.0f) ? m : 0.2f * m;
    partial += lr * sAtt[c0 + j];
  }
  partial += __shfl_xor(partial, 1, 32);
  partial += __shfl_xor(partial, 2, 32);
  partial += __shfl_xor(partial, 4, 32);
  if ((lane & 7) == 0) {
    const int h = lane >> 3;
    logits[(size_t)e * GH + h] = partial;
    atomicMaxF(mx + (size_t)d * GH + h, partial);
  }
}

__global__ void edge_exp_kernel(const float* __restrict__ logits,
                                const float* __restrict__ mx,
                                float* __restrict__ den, float* __restrict__ exb,
                                const int* __restrict__ didx, int E)
{
  const int idx = blockIdx.x * blockDim.x + threadIdx.x;
  if (idx >= E * GH) return;
  const int e = idx >> 2, h = idx & 3;
  const int d = didx[e];
  const float v = __expf(logits[idx] - mx[(size_t)d * GH + h]);
  exb[idx] = v;
  atomicAdd(den + (size_t)d * GH + h, v);
}

__global__ __launch_bounds__(256)
void edge_scatter_kernel(const float* __restrict__ xl, const float* __restrict__ exb,
                         const float* __restrict__ den,
                         const int* __restrict__ sidx, const int* __restrict__ didx,
                         float* __restrict__ acc, int E)
{
  const int lane = threadIdx.x & 31;
  const int e = blockIdx.x * 8 + (threadIdx.x >> 5);
  if (e >= E) return;
  const int s = sidx[e], d = didx[e];
  const int h = lane >> 3;
  const float alpha = exb[(size_t)e * GH + h] / den[(size_t)d * GH + h];
  const int c0 = lane * 8;
  const float4* lp = (const float4*)(xl + (size_t)s * GD + c0);
  float4 l0 = lp[0], l1 = lp[1];
  float v[8] = {l0.x, l0.y, l0.z, l0.w, l1.x, l1.y, l1.z, l1.w};
  float* ap = acc + (size_t)d * GD + c0;
#pragma unroll
  for (int j = 0; j < 8; ++j) atomicAdd(ap + j, alpha * v[j]);
}

__global__ void finalize_gat_kernel(const float* __restrict__ acc,
                                    const float* __restrict__ bias,
                                    float* __restrict__ out, int N)
{
  const int idx = blockIdx.x * blockDim.x + threadIdx.x;
  if (idx >= N * HC) return;
  const int n = idx >> 6, c = idx & 63;
  const float* a = acc + (size_t)n * GD;
  const float v = (a[c] + a[HC + c] + a[2 * HC + c] + a[3 * HC + c]) * 0.25f + bias[c];
  out[idx] = gelu_exact(v);
}

// ---------------------------------------------------------------------------
__global__ void colsum_kernel(const float* __restrict__ X, float* __restrict__ out,
                              int R, int C)
{
  const int gt = blockIdx.x * blockDim.x + threadIdx.x;
  const int total = gridDim.x * blockDim.x;      // multiple of C
  const int c = gt % C;
  const int r0 = gt / C;
  const int rs = total / C;
  float acc = 0.0f;
  for (int r = r0; r < R; r += rs) acc += X[(size_t)r * C + c];
  atomicAdd(out + c, acc);
}

__global__ void state_proj_kernel(const float* __restrict__ sums,
                                  const float* __restrict__ W,
                                  const float* __restrict__ b,
                                  float* __restrict__ state)
{
  __shared__ float gs[144];
  const int t = threadIdx.x;
  if (t < 144) {
    const float scale = (t < 64) ? (1.0f / NP) : (t < 128 ? (1.0f / NB) : (1.0f / NE));
    gs[t] = sums[t] * scale;
  }
  __syncthreads();
  if (t < MD) {
    const float* w = W + (size_t)t * 144;
    float acc = b[t];
    for (int j = 0; j < 144; ++j) acc += gs[j] * w[j];
    state[t] = acc;
  }
}

// ---------------------------------------------------------------------------
// Tiny transformer: seq len 6, dim 128, 2 post-norm layers, 4 heads, GELU FFN.
// Single workgroup, everything in LDS. (~2 MFLOP; WMMA would be pure overhead.)
// ---------------------------------------------------------------------------
struct TfP {
  const float *Wqkv, *bqkv, *Wo, *bo, *ln1w, *ln1b, *ln2w, *ln2b, *W1, *b1, *W2, *b2;
};

__global__ __launch_bounds__(256)
void transformer_kernel(const float* __restrict__ history,
                        const float* __restrict__ state,
                        const float* __restrict__ pos,
                        TfP l0, TfP l1, float* __restrict__ ctx)
{
  __shared__ float x[6 * MD];
  __shared__ float y[6 * MD];
  __shared__ float qkv[6 * 3 * MD];
  __shared__ float ao[6 * MD];
  __shared__ float ff[6 * 4 * MD];
  __shared__ float sc[4 * 6 * 6];
  __shared__ float mu[6], rs[6];
  const int tid = threadIdx.x;

  for (int t = tid; t < 6 * MD; t += 256) {
    const int s = t >> 7, j = t & 127;
    x[t] = ((s < LBH) ? history[s * MD + j] : state[j]) + pos[t];
  }
  __syncthreads();

  const TfP L[2] = {l0, l1};
#pragma unroll 1
  for (int li = 0; li < 2; ++li) {
    const TfP p = L[li];
    for (int t = tid; t < 6 * 384; t += 256) {
      const int s = t / 384, o = t - s * 384;
      const float* w = p.Wqkv + (size_t)o * MD;
      float a = p.bqkv[o];
      for (int j = 0; j < MD; ++j) a += x[s * MD + j] * w[j];
      qkv[t] = a;
    }
    __syncthreads();
    for (int t = tid; t < 144; t += 256) {
      const int h = t / 36, rem = t - h * 36, s = rem / 6, u = rem - s * 6;
      float a = 0.0f;
      for (int k = 0; k < 32; ++k)
        a += qkv[s * 384 + h * 32 + k] * qkv[u * 384 + MD + h * 32 + k];
      sc[(h * 6 + s) * 6 + u] = a * 0.17677669529663687f;   // 1/sqrt(32)
    }
    __syncthreads();
    for (int t = tid; t < 24; t += 256) {
      float* row = sc + t * 6;
      float m = row[0];
      for (int u = 1; u < 6; ++u) m = fmaxf(m, row[u]);
      float ssum = 0.0f;
      for (int u = 0; u < 6; ++u) { const float e2 = __expf(row[u] - m); row[u] = e2; ssum += e2; }
      const float inv = 1.0f / ssum;
      for (int u = 0; u < 6; ++u) row[u] *= inv;
    }
    __syncthreads();
    for (int t = tid; t < 6 * MD; t += 256) {
      const int s = t >> 7, j = t & 127, h = j >> 5;
      float a = 0.0f;
      for (int u = 0; u < 6; ++u)
        a += sc[(h * 6 + s) * 6 + u] * qkv[u * 384 + 2 * MD + j];
      ao[t] = a;
    }
    __syncthreads();
    for (int t = tid; t < 6 * MD; t += 256) {
      const int s = t >> 7, o = t & 127;
      const float* w = p.Wo + (size_t)o * MD;
      float a = p.bo[o];
      for (int j = 0; j < MD; ++j) a += ao[s * MD + j] * w[j];
      y[t] = x[t] + a;
    }
    __syncthreads();
    for (int t = tid; t < 6; t += 256) {
      float m = 0.0f;
      for (int j = 0; j < MD; ++j) m += y[t * MD + j];
      m /= (float)MD;
      float v = 0.0f;
      for (int j = 0; j < MD; ++j) { const float d = y[t * MD + j] - m; v += d * d; }
      mu[t] = m; rs[t] = rsqrtf(v / (float)MD + 1e-5f);
    }
    __syncthreads();
    for (int t = tid; t < 6 * MD; t += 256) {
      const int s = t >> 7, j = t & 127;
      x[t] = (y[t] - mu[s]) * rs[s] * p.ln1w[j] + p.ln1b[j];
    }
    __syncthreads();
    for (int t = tid; t < 6 * 4 * MD; t += 256) {
      const int s = t / 512, o = t - s * 512;
      const float* w = p.W1 + (size_t)o * MD;
      float a = p.b1[o];
      for (int j = 0; j < MD; ++j) a += x[s * MD + j] * w[j];
      ff[t] = gelu_exact(a);
    }
    __syncthreads();
    for (int t = tid; t < 6 * MD; t += 256) {
      const int s = t >> 7, o = t & 127;
      const float* w = p.W2 + (size_t)o * 512;
      float a = p.b2[o];
      for (int j = 0; j < 512; ++j) a += ff[s * 512 + j] * w[j];
      y[t] = x[t] + a;
    }
    __syncthreads();
    for (int t = tid; t < 6; t += 256) {
      float m = 0.0f;
      for (int j = 0; j < MD; ++j) m += y[t * MD + j];
      m /= (float)MD;
      float v = 0.0f;
      for (int j = 0; j < MD; ++j) { const float d = y[t * MD + j] - m; v += d * d; }
      mu[t] = m; rs[t] = rsqrtf(v / (float)MD + 1e-5f);
    }
    __syncthreads();
    for (int t = tid; t < 6 * MD; t += 256) {
      const int s = t >> 7, j = t & 127;
      x[t] = (y[t] - mu[s]) * rs[s] * p.ln2w[j] + p.ln2b[j];
    }
    __syncthreads();
  }
  for (int t = tid; t < MD; t += 256) ctx[t] = x[5 * MD + t];
}

// ---------------------------------------------------------------------------
__global__ void gather_tok_kernel(const float* __restrict__ hP, const float* __restrict__ hB,
                                  const float* __restrict__ ea,
                                  const int* __restrict__ pidx, const int* __restrict__ bidx,
                                  const int* __restrict__ eidx, __bf16* __restrict__ tok)
{
  const int idx = blockIdx.x * blockDim.x + threadIdx.x;
  if (idx >= NS * 144) return;
  const int i = idx / 144, j = idx - i * 144;
  float v;
  if (j < 64)       v = hP[(size_t)pidx[i] * 64 + j];
  else if (j < 128) v = hB[(size_t)bidx[i] * 64 + (j - 64)];
  else              v = ea[(size_t)eidx[i] * 16 + (j - 128)];
  tok[idx] = f2bf(v);
}

__global__ void bcast_ctx_kernel(const float* __restrict__ ctx, __bf16* __restrict__ z)
{
  const int idx = blockIdx.x * blockDim.x + threadIdx.x;
  if (idx >= NS * MD) return;
  const int i = idx >> 7, j = idx & 127;
  z[(size_t)i * 256 + MD + j] = f2bf(ctx[j]);
}

__global__ void cls2_kernel(const float* __restrict__ h1, const float* __restrict__ W2,
                            const float* __restrict__ b2, float* __restrict__ out)
{
  const int idx = blockIdx.x * blockDim.x + threadIdx.x;
  if (idx >= NS * NCL) return;
  const int i = idx >> 3, o = idx & 7;
  const float* hv = h1 + (size_t)i * MD;
  const float* w  = W2 + (size_t)o * MD;
  float acc = b2[o];
#pragma unroll 4
  for (int j = 0; j < MD; ++j) acc += hv[j] * w[j];
  out[idx] = acc;
}

// ---------------------------------------------------------------------------
extern "C" void kernel_launch(void* const* d_in, const int* in_sizes, int n_in,
                              void* d_out, int out_size, void* d_ws, size_t ws_size,
                              hipStream_t stream)
{
  (void)in_sizes; (void)n_in; (void)out_size; (void)ws_size;
  auto F = [&](int i) { return (const float*)d_in[i]; };

  // inputs (setup_inputs order; params pytree in sorted-key order)
  const float* pitcher_x = F(0);
  const float* batter_x  = F(1);
  const float* ea        = F(2);
  const float* history   = F(3);
  const float* bpW = F(4),  *bpB = F(5);
  const float* cW1 = F(6),  *cW2 = F(7),  *cb1 = F(8), *cb2 = F(9);
  const float* etW = F(10), *etB = F(11);
  const float* fWe = F(12), *fWl = F(13), *fWr = F(14), *fAtt = F(15),
             * fBias = F(16), *fbl = F(17), *fbr = F(18);
  const float* rWe = F(19), *rWl = F(20), *rWr = F(21), *rAtt = F(22),
             * rBias = F(23), *rbl = F(24), *rbr = F(25);
  const float* gsW = F(26), *gsB = F(27);
  TfP l0 = {F(31), F(35), F(30), F(34), F(37), F(36), F(39), F(38), F(28), F(32), F(29), F(33)};
  TfP l1 = {F(43), F(47), F(42), F(46), F(49), F(48), F(51), F(50), F(40), F(44), F(41), F(45)};
  const float* ppW = F(52), *ppB = F(53);
  const float* pos = F(54);
  const int* eindex = (const int*)d_in[55];
  const int* src = eindex;            // pitcher ids
  const int* dst = eindex + NE;       // batter ids
  const int* pidx = (const int*)d_in[56];
  const int* bidx = (const int*)d_in[57];
  const int* eidx = (const int*)d_in[58];

  // workspace layout (bytes, 256B-aligned chunks)
  char* ws = (char*)d_ws;
  size_t off = 0;
  auto allocB = [&](size_t bytes) { char* p = ws + off; off += (bytes + 255) & ~(size_t)255; return p; };
  auto aF  = [&](size_t n) { return (float*)allocB(n * 4); };
  auto aBF = [&](size_t n) { return (__bf16*)allocB(n * 2); };

  // bf16 operand pool (GEMM inputs)
  __bf16* px_bf = aBF((size_t)NP * 64);
  __bf16* bx_bf = aBF((size_t)NB * 64);
  __bf16* hp_bf = aBF((size_t)NP * 64);
  __bf16* hb_bf = aBF((size_t)NB * 64);
  __bf16* ppW_bf = aBF(64 * 64);
  __bf16* bpW_bf = aBF(64 * 64);
  __bf16* fWl_bf = aBF(GD * 64);
  __bf16* fWr_bf = aBF(GD * 64);
  __bf16* rWl_bf = aBF(GD * 64);
  __bf16* rWr_bf = aBF(GD * 64);
  __bf16* etW_bf = aBF(MD * 144);
  __bf16* cW1_bf = aBF(MD * 256);
  __bf16* tok_bf = aBF((size_t)NS * 144);
  __bf16* zb_bf  = aBF((size_t)NS * 256);
  // f32 buffers
  float* hP   = aF((size_t)NP * HC);
  float* hB   = aF((size_t)NB * HC);
  float* xl   = aF((size_t)NB * GD);
  float* xr   = aF((size_t)NB * GD);
  float* accb = aF((size_t)NB * GD);
  float* lg   = aF((size_t)NE * GH);
  float* exb  = aF((size_t)NE * GH);
  float* mx   = aF((size_t)NB * GH);
  float* den  = aF((size_t)NB * GH);
  float* sums = aF(160);               // [0:64] h_p, [64:128] h_b, [128:144] ea
  float* state= aF(MD);
  float* ctx  = aF(MD);
  float* h1   = aF((size_t)NS * MD);

  auto cvt = [&](const float* in, __bf16* out, long long n) {
    f32_to_bf16_kernel<<<(unsigned)((n + 255) / 256), 256, 0, stream>>>(in, out, n);
  };
  auto gemmF = [&](const __bf16* A, const __bf16* W, const float* b, float* C,
                   int M, int N, int K, int ldc, int act) {
    const int tiles = ((M + 15) / 16) * (N / 16);
    const int blocks = (tiles + 7) / 8;
    if ((K & 31) == 0)
      gemm_wmma_bf16<true><<<blocks, 256, 0, stream>>>(A, W, b, C, (__bf16*)nullptr, M, N, K, ldc, act);
    else
      gemm_wmma_bf16<false><<<blocks, 256, 0, stream>>>(A, W, b, C, (__bf16*)nullptr, M, N, K, ldc, act);
  };
  auto gemmB = [&](const __bf16* A, const __bf16* W, const float* b, __bf16* C,
                   int M, int N, int K, int ldc, int act) {
    const int tiles = ((M + 15) / 16) * (N / 16);
    const int blocks = (tiles + 7) / 8;
    if ((K & 31) == 0)
      gemm_wmma_bf16<true><<<blocks, 256, 0, stream>>>(A, W, b, (float*)nullptr, C, M, N, K, ldc, act);
    else
      gemm_wmma_bf16<false><<<blocks, 256, 0, stream>>>(A, W, b, (float*)nullptr, C, M, N, K, ldc, act);
  };
  auto fill = [&](float* p, float v, long long n) {
    fill_kernel<<<(unsigned)((n + 255) / 256), 256, 0, stream>>>(p, v, n);
  };

  // 0) one-time bf16 conversions (inputs + GEMM weights)
  cvt(pitcher_x, px_bf, (long long)NP * 64);
  cvt(batter_x,  bx_bf, (long long)NB * 64);
  cvt(ppW, ppW_bf, 64 * 64);
  cvt(bpW, bpW_bf, 64 * 64);
  cvt(fWl, fWl_bf, GD * 64);
  cvt(fWr, fWr_bf, GD * 64);
  cvt(rWl, rWl_bf, GD * 64);
  cvt(rWr, rWr_bf, GD * 64);
  cvt(etW, etW_bf, MD * 144);
  cvt(cW1, cW1_bf, MD * 256);

  // 1) node projections (gelu), outputs stay bf16 (only ever GEMM A operands)
  gemmB(px_bf, ppW_bf, ppB, hp_bf, NP, 64, 64, 64, 1);
  gemmB(bx_bf, bpW_bf, bpB, hb_bf, NB, 64, 64, 64, 1);

  // 2) GATv2 forward (pitcher -> batter), segments over batters
  gemmF(hp_bf, fWl_bf, fbl, xl, NP, GD, 64, GD, 0);
  gemmF(hb_bf, fWr_bf, fbr, xr, NB, GD, 64, GD, 0);
  fill(mx, -INFINITY, (long long)NB * GH);
  fill(den, 0.0f, (long long)NB * GH);
  fill(accb, 0.0f, (long long)NB * GD);
  edge_logits_kernel<<<(NE + 7) / 8, 256, 0, stream>>>(xl, xr, ea, fWe, fAtt, src, dst, lg, mx, NE);
  edge_exp_kernel<<<(NE * GH + 255) / 256, 256, 0, stream>>>(lg, mx, den, exb, dst, NE);
  edge_scatter_kernel<<<(NE + 7) / 8, 256, 0, stream>>>(xl, exb, den, src, dst, accb, NE);
  finalize_gat_kernel<<<(NB * HC + 255) / 256, 256, 0, stream>>>(accb, fBias, hB, NB);

  // 3) GATv2 reverse (batter -> pitcher), segments over pitchers (buffers reused)
  gemmF(hb_bf, rWl_bf, rbl, xl, NB, GD, 64, GD, 0);
  gemmF(hp_bf, rWr_bf, rbr, xr, NP, GD, 64, GD, 0);
  fill(mx, -INFINITY, (long long)NP * GH);
  fill(den, 0.0f, (long long)NP * GH);
  fill(accb, 0.0f, (long long)NP * GD);
  edge_logits_kernel<<<(NE + 7) / 8, 256, 0, stream>>>(xl, xr, ea, rWe, rAtt, dst, src, lg, mx, NE);
  edge_exp_kernel<<<(NE * GH + 255) / 256, 256, 0, stream>>>(lg, mx, den, exb, src, NE);
  edge_scatter_kernel<<<(NE + 7) / 8, 256, 0, stream>>>(xl, exb, den, dst, src, accb, NE);
  finalize_gat_kernel<<<(NP * HC + 255) / 256, 256, 0, stream>>>(accb, rBias, hP, NP);

  // 4) graph state
  fill(sums, 0.0f, 160);
  colsum_kernel<<<128, 256, 0, stream>>>(hP, sums, NP, 64);
  colsum_kernel<<<128, 256, 0, stream>>>(hB, sums + 64, NB, 64);
  colsum_kernel<<<128, 256, 0, stream>>>(ea, sums + 128, NE, 16);
  state_proj_kernel<<<1, 192, 0, stream>>>(sums, gsW, gsB, state);

  // 5) temporal transformer -> ctx
  transformer_kernel<<<1, 256, 0, stream>>>(history, state, pos, l0, l1, ctx);

  // 6) token path + classifier
  gather_tok_kernel<<<(NS * 144 + 255) / 256, 256, 0, stream>>>(hP, hB, ea, pidx, bidx, eidx, tok_bf);
  gemmB(tok_bf, etW_bf, etB, zb_bf, NS, MD, 144, 256, 0);                // z[:,0:128]
  bcast_ctx_kernel<<<(NS * MD + 255) / 256, 256, 0, stream>>>(ctx, zb_bf); // z[:,128:256]
  gemmF(zb_bf, cW1_bf, cb1, h1, NS, MD, 256, MD, 1);                     // gelu
  cls2_kernel<<<(NS * NCL + 255) / 256, 256, 0, stream>>>(h1, cW2, cb2, (float*)d_out);
}
